// EquiMAB3_1322849927409
// MI455X (gfx1250) — matrix operand order repaired and checked by reading, not compiled, not run on hardware
//
#include <hip/hip_runtime.h>
#include <hip/hip_bf16.h>

// Problem constants (from reference): B=8, N=M=512, S=4, D=L=256, H=4, dh=64.
#define NB   8
#define NN   512
#define NM   512
#define NS   4
#define ND   256
#define NL   256
#define NH   4
#define ROWS 16384   // B*N*S

typedef __attribute__((ext_vector_type(16))) __bf16 v16bf;
typedef __attribute__((ext_vector_type(8)))  float  v8f;

// scheduler pipeline directives (best-effort groups)
#define SG_WMMA(n)   __builtin_amdgcn_sched_group_barrier(0x008, (n), 0)
#define SG_DSREAD(n) __builtin_amdgcn_sched_group_barrier(0x100, (n), 0)

union Frag {
    v16bf  v;
    __bf16 e[16];
    uint4  q[2];
};

// ---- WMMA fragment loaders (bf16 16x16x32) ----------------------------------
// A-matrix 16x32 per-lane element map: K = (e&8)*2 + lhi*8 + (e&7), M = lane&15.
// => two contiguous 16B chunks at element offsets lhi*8 and 16+lhi*8.
__device__ __forceinline__ v16bf load_frag_a(const __bf16* rowK32, int lhi) {
    Frag f;
    f.q[0] = *(const uint4*)(rowK32 + lhi * 8);
    f.q[1] = *(const uint4*)(rowK32 + 16 + lhi * 8);
    return f.v;
}
// B-matrix 32x16, staged *transposed* in LDS as [n][k]: lane n=lane&15 reads
// K = (e&8)*2 + lhi*8 + (e&7)  => two 16B chunks at lhi*8 and 16+lhi*8.
__device__ __forceinline__ v16bf load_frag_b(const __bf16* rowK32, int lhi) {
    Frag f;
    f.q[0] = *(const uint4*)(rowK32 + lhi * 8);
    f.q[1] = *(const uint4*)(rowK32 + 16 + lhi * 8);
    return f.v;
}

__device__ __forceinline__ v8f wmma_bf16(v16bf a, v16bf b, v8f c) {
    return __builtin_amdgcn_wmma_f32_16x16x32_bf16(false, a, false, b,
                                                   (short)0, c, false, false);
}

__device__ __forceinline__ v8f vzero8() {
    v8f z;
#pragma unroll
    for (int e = 0; e < 8; ++e) z[e] = 0.f;
    return z;
}

// ---- cross-lane reductions within 16-lane halves (matches C/D layout) -------
template <int MASK>
__device__ __forceinline__ float swz_xor(float v) {
    int i = __builtin_amdgcn_ds_swizzle(__float_as_int(v), 0x1f | (MASK << 10));
    return __int_as_float(i);
}
__device__ __forceinline__ float hred_max(float v) {
    v = fmaxf(v, swz_xor<1>(v)); v = fmaxf(v, swz_xor<2>(v));
    v = fmaxf(v, swz_xor<4>(v)); v = fmaxf(v, swz_xor<8>(v));
    return v;
}
__device__ __forceinline__ float hred_sum(float v) {
    v += swz_xor<1>(v); v += swz_xor<2>(v);
    v += swz_xor<4>(v); v += swz_xor<8>(v);
    return v;
}

// =============================================================================
// Kernel 1: projections.  out = X @ W^T + b for (Q,Wq),(K,Wk),(K,Wv).
// Writes Qp (f32, for residual) and head-major bf16 buffers [H][B][N][S*dh].
// =============================================================================
__global__ void __launch_bounds__(256, 1)
proj_kernel(const float* __restrict__ Qin,
            const float* __restrict__ Kin,
            const float* __restrict__ Wq, const float* __restrict__ bq,
            const float* __restrict__ Wk, const float* __restrict__ bk,
            const float* __restrict__ Wv, const float* __restrict__ bv,
            float*  __restrict__ Qp,
            __bf16* __restrict__ Qhb,
            __bf16* __restrict__ Khb,
            __bf16* __restrict__ Vhb) {
    extern __shared__ char smem[];
    __bf16* ldsX = (__bf16*)smem;     // [128][32]
    __bf16* ldsW = ldsX + 128 * 32;   // [256][32]  (layout [n][k] = W rows)

    const int p = blockIdx.y;
    const float* src  = (p == 0) ? Qin : Kin;
    const float* W    = (p == 0) ? Wq : (p == 1) ? Wk : Wv;
    const float* bias = (p == 0) ? bq : (p == 1) ? bk : bv;
    __bf16* dsth      = (p == 0) ? Qhb : (p == 1) ? Khb : Vhb;

    const int row0 = blockIdx.x * 128;
    const int tid  = threadIdx.x;
    const int lane = tid & 31;
    const int wave = tid >> 5;
    const int lhi  = lane >> 4;
    const int ln   = lane & 15;

    v8f acc[16];
#pragma unroll
    for (int t = 0; t < 16; ++t) acc[t] = vzero8();

    for (int kb = 0; kb < 8; ++kb) {
        __syncthreads();
        // stage X slab 128x32 (f32 -> bf16)
        for (int i = tid; i < 128 * 8; i += 256) {
            int r = i >> 3, c4 = (i & 7) << 2;
            float4 f = *(const float4*)(src + (size_t)(row0 + r) * ND + kb * 32 + c4);
            __bf16* d = ldsX + r * 32 + c4;
            d[0] = (__bf16)f.x; d[1] = (__bf16)f.y; d[2] = (__bf16)f.z; d[3] = (__bf16)f.w;
        }
        // stage W slab 256x32
        for (int i = tid; i < 256 * 8; i += 256) {
            int n = i >> 3, c4 = (i & 7) << 2;
            float4 f = *(const float4*)(W + (size_t)n * ND + kb * 32 + c4);
            __bf16* d = ldsW + n * 32 + c4;
            d[0] = (__bf16)f.x; d[1] = (__bf16)f.y; d[2] = (__bf16)f.z; d[3] = (__bf16)f.w;
        }
        __syncthreads();

        v16bf af = load_frag_a(ldsX + (wave * 16 + ln) * 32, lhi);
        v16bf bf = load_frag_b(ldsW + ln * 32, lhi);
#pragma unroll
        for (int ct = 0; ct < 16; ++ct) {
            v16bf cur = bf;
            if (ct + 1 < 16)
                bf = load_frag_b(ldsW + ((ct + 1) * 16 + ln) * 32, lhi);
            acc[ct] = wmma_bf16(af, cur, acc[ct]);
        }
        // pipeline shape: 10 DS reads up front, then {1 WMMA, 4 DS reads} x14,
        // tail 2 WMMA  (totals: 66 DS reads, 16 WMMA)
        SG_DSREAD(10);
#pragma unroll
        for (int i = 0; i < 14; ++i) { SG_WMMA(1); SG_DSREAD(4); }
        SG_WMMA(2);
    }

    // epilogue: bias, store f32 Qp + bf16 head-major
#pragma unroll
    for (int ct = 0; ct < 16; ++ct) {
        int col = ct * 16 + ln;
        float bv_ = bias[col];
        int hh = col >> 6, dd = col & 63;
#pragma unroll
        for (int vr = 0; vr < 8; ++vr) {
            int r = row0 + wave * 16 + vr + lhi * 8;   // r = (b*N+n)*S + s
            float val = acc[ct][vr] + bv_;
            if (p == 0) Qp[(size_t)r * NL + col] = val;
            int s = r & 3, bn = r >> 2;                 // S == 4
            size_t hidx = (((size_t)(hh * NB + (bn >> 9)) * NN) + (bn & 511)) * 256
                        + s * 64 + dd;
            dsth[hidx] = (__bf16)val;
        }
    }
}

// =============================================================================
// Kernel 2: flash attention per (h,b).  E = Qh Kh^T / 16, softmax over M,
// O = Qh + A Vh.  Writes Xattn in [B,N,S,L] f32.
// =============================================================================
__global__ void __launch_bounds__(256, 1)
attn_kernel(const __bf16* __restrict__ Qhb,
            const __bf16* __restrict__ Khb,
            const __bf16* __restrict__ Vhb,
            const float*  __restrict__ Qp,
            const int*    __restrict__ mask,
            float*        __restrict__ Xattn) {
    extern __shared__ char smem[];
    __bf16* ldsQ  = (__bf16*)smem;      // [128][256]
    __bf16* ldsK  = ldsQ + 128 * 256;   // [32][256]
    __bf16* ldsVt = ldsK + 32 * 256;    // [256][32]  (transposed V block)
    __bf16* ldsP  = ldsVt + 256 * 32;   // [8 waves][16][32]

    const int hb = blockIdx.y;
    const int h = hb & 3, b = hb >> 2;
    const int q0 = blockIdx.x * 128;
    const int tid  = threadIdx.x;
    const int lane = tid & 31;
    const int wave = tid >> 5;
    const int lhi  = lane >> 4;
    const int ln   = lane & 15;

    const size_t head = (size_t)(h * NB + b) * NN * 256;

    // stage Q block once (128x256 bf16 = 64KB)
    {
        const uint4* s = (const uint4*)(Qhb + head + (size_t)q0 * 256);
        uint4* d = (uint4*)ldsQ;
        for (int i = tid; i < 4096; i += 256) d[i] = s[i];
    }

    float mrun[8], lrun[8];
    v8f Oa[16];
#pragma unroll
    for (int vr = 0; vr < 8; ++vr) { mrun[vr] = -1e30f; lrun[vr] = 0.f; }
#pragma unroll
    for (int t = 0; t < 16; ++t) Oa[t] = vzero8();

    const __bf16* qrow = ldsQ + (wave * 16 + ln) * 256;

    for (int kb = 0; kb < 16; ++kb) {   // 32 keys per block
        __syncthreads();
        {   // stage K block [32][256]
            const uint4* s = (const uint4*)(Khb + head + (size_t)(kb * 32) * 256);
            uint4* d = (uint4*)ldsK;
            for (int i = tid; i < 1024; i += 256) d[i] = s[i];
        }
        {   // stage V block transposed -> [sd][m]
            const __bf16* s = Vhb + head + (size_t)(kb * 32) * 256;
            for (int i = tid; i < 8192; i += 256) {
                int m = i >> 8, sd = i & 255;
                ldsVt[sd * 32 + m] = s[m * 256 + sd];
            }
        }
        __syncthreads();

        // E tiles: 16 q-rows x 32 keys, contraction over 256 (pipelined)
        v8f E[2];
        E[0] = vzero8(); E[1] = vzero8();
        v16bf af = load_frag_a(qrow, lhi);
        v16bf b0 = load_frag_b(ldsK + ln * 256, lhi);
        v16bf b1 = load_frag_b(ldsK + (16 + ln) * 256, lhi);
#pragma unroll
        for (int ks = 0; ks < 8; ++ks) {
            v16bf ca = af, c0 = b0, c1 = b1;
            if (ks + 1 < 8) {
                af = load_frag_a(qrow + (ks + 1) * 32, lhi);
                b0 = load_frag_b(ldsK + ln * 256 + (ks + 1) * 32, lhi);
                b1 = load_frag_b(ldsK + (16 + ln) * 256 + (ks + 1) * 32, lhi);
            }
            E[0] = wmma_bf16(ca, c0, E[0]);
            E[1] = wmma_bf16(ca, c1, E[1]);
        }
        // pipeline shape: 48 DS reads total, 16 WMMA
        SG_DSREAD(12);
#pragma unroll
        for (int i = 0; i < 6; ++i) { SG_WMMA(2); SG_DSREAD(6); }
        SG_WMMA(4);

        // scale + mask
        const float scale = 0.0625f;   // 1/sqrt(L) = 1/16
#pragma unroll
        for (int t = 0; t < 2; ++t) {
            int mg = kb * 32 + t * 16 + ln;
#pragma unroll
            for (int vr = 0; vr < 8; ++vr) {
                int qg = q0 + wave * 16 + vr + lhi * 8;
                float e = E[t][vr] * scale;
                if (mask[(size_t)qg * NM + mg] == 0) e = -1e30f;
                E[t][vr] = e;
            }
        }

        // online softmax
        float cf[8];
#pragma unroll
        for (int vr = 0; vr < 8; ++vr) {
            float rm = hred_max(fmaxf(E[0][vr], E[1][vr]));
            float mnew = fmaxf(mrun[vr], rm);
            cf[vr] = __expf(mrun[vr] - mnew);
            mrun[vr] = mnew;
            float p0 = __expf(E[0][vr] - mnew);
            float p1 = __expf(E[1][vr] - mnew);
            E[0][vr] = p0; E[1][vr] = p1;
            lrun[vr] = lrun[vr] * cf[vr] + hred_sum(p0 + p1);
        }

        // P -> per-wave LDS in A-fragment-friendly [q][key32] layout
        __bf16* myP = ldsP + wave * 16 * 32;
#pragma unroll
        for (int t = 0; t < 2; ++t)
#pragma unroll
            for (int vr = 0; vr < 8; ++vr)
                myP[(vr + lhi * 8) * 32 + t * 16 + ln] = (__bf16)E[t][vr];
        asm volatile("s_wait_dscnt 0" ::: "memory");   // per-wave LDS RAW
        v16bf pf = load_frag_a(myP + ln * 32, lhi);

        // rescale O and accumulate P @ V (pipelined B fragments)
        v16bf vf = load_frag_b(ldsVt + ln * 32, lhi);
#pragma unroll
        for (int sdt = 0; sdt < 16; ++sdt) {
            v16bf cv = vf;
            if (sdt + 1 < 16)
                vf = load_frag_b(ldsVt + ((sdt + 1) * 16 + ln) * 32, lhi);
#pragma unroll
            for (int vr = 0; vr < 8; ++vr) Oa[sdt][vr] *= cf[vr];
            Oa[sdt] = wmma_bf16(pf, cv, Oa[sdt]);
        }
        // pipeline shape: 34 DS reads (pf 2 + 16x2), 16 WMMA
        SG_DSREAD(4);
#pragma unroll
        for (int i = 0; i < 15; ++i) { SG_WMMA(1); SG_DSREAD(2); }
        SG_WMMA(1);
    }

    // epilogue: normalize, residual with Qh (from f32 Qp), scatter to [B,N,S,L]
#pragma unroll
    for (int vr = 0; vr < 8; ++vr)
        lrun[vr] = (lrun[vr] > 0.f) ? (1.f / lrun[vr]) : 0.f;
#pragma unroll
    for (int sdt = 0; sdt < 16; ++sdt) {
        int sd = sdt * 16 + ln;
        int s = sd >> 6, dd = sd & 63;
#pragma unroll
        for (int vr = 0; vr < 8; ++vr) {
            int n = q0 + wave * 16 + vr + lhi * 8;
            size_t gi = (((size_t)(b * NN + n)) * NS + s) * NL + h * 64 + dd;
            Xattn[gi] = Oa[sdt][vr] * lrun[vr] + Qp[gi];
        }
    }
}

// =============================================================================
// Kernel 3: y = LN0(x); z = y + relu(y @ Wo^T + bo); out = LN1(z).
// =============================================================================
__global__ void __launch_bounds__(256, 1)
mlp_ln_kernel(const float* __restrict__ X,
              const float* __restrict__ Wo, const float* __restrict__ bo,
              const float* __restrict__ g0, const float* __restrict__ b0,
              const float* __restrict__ g1, const float* __restrict__ b1,
              float* __restrict__ out) {
    extern __shared__ char smem[];
    float*  ldsY   = (float*)smem;                   // [128][256] f32
    __bf16* ldsW   = (__bf16*)(ldsY + 128 * 256);    // [256][32]
    float*  ldsPar = (float*)(ldsW + 256 * 32);      // [5][256]: g0,b0,bo,g1,b1

    const int row0 = blockIdx.x * 128;
    const int tid  = threadIdx.x;
    const int lane = tid & 31;
    const int wave = tid >> 5;
    const int lhi  = lane >> 4;
    const int ln   = lane & 15;

    ldsPar[0 * 256 + tid] = g0[tid];
    ldsPar[1 * 256 + tid] = b0[tid];
    ldsPar[2 * 256 + tid] = bo[tid];
    ldsPar[3 * 256 + tid] = g1[tid];
    ldsPar[4 * 256 + tid] = b1[tid];
    __syncthreads();

    // ---- phase 1: LN0 into ldsY (2 threads per row) ----
    {
        int r = tid >> 1, hf = tid & 1;
        const float* srow = X + (size_t)(row0 + r) * NL + hf * 128;
        float* drow = ldsY + r * 256 + hf * 128;
        float s = 0.f, s2 = 0.f;
        for (int c = 0; c < 128; c += 4) {
            float4 f = *(const float4*)(srow + c);
            *(float4*)(drow + c) = f;
            s  += f.x + f.y + f.z + f.w;
            s2 += f.x * f.x + f.y * f.y + f.z * f.z + f.w * f.w;
        }
        s  += swz_xor<1>(s);
        s2 += swz_xor<1>(s2);
        float mu  = s * (1.f / 256.f);
        float var = s2 * (1.f / 256.f) - mu * mu;
        float rsig = rsqrtf(var + 1e-5f);
        for (int c = 0; c < 128; ++c) {
            int col = hf * 128 + c;
            drow[c] = (drow[c] - mu) * rsig * ldsPar[col] + ldsPar[256 + col];
        }
    }

    // ---- phase 2: z_acc = y @ Wo^T via WMMA ----
    v8f acc[16];
#pragma unroll
    for (int t = 0; t < 16; ++t) acc[t] = vzero8();

    for (int kb = 0; kb < 8; ++kb) {
        __syncthreads();
        for (int i = tid; i < 256 * 8; i += 256) {
            int n = i >> 3, c4 = (i & 7) << 2;
            float4 f = *(const float4*)(Wo + (size_t)n * NL + kb * 32 + c4);
            __bf16* d = ldsW + n * 32 + c4;
            d[0] = (__bf16)f.x; d[1] = (__bf16)f.y; d[2] = (__bf16)f.z; d[3] = (__bf16)f.w;
        }
        __syncthreads();

        // A fragment: f32 LDS -> bf16 on the fly
        Frag af;
        const float* yrow = ldsY + (wave * 16 + ln) * 256 + kb * 32;
#pragma unroll
        for (int e = 0; e < 8; ++e) af.e[e]     = (__bf16)yrow[lhi * 8 + e];
#pragma unroll
        for (int e = 0; e < 8; ++e) af.e[8 + e] = (__bf16)yrow[16 + lhi * 8 + e];

        // double-buffered B fragments
        v16bf bf = load_frag_b(ldsW + ln * 32, lhi);
#pragma unroll
        for (int ct = 0; ct < 16; ++ct) {
            v16bf cur = bf;
            if (ct + 1 < 16)
                bf = load_frag_b(ldsW + ((ct + 1) * 16 + ln) * 32, lhi);
            acc[ct] = wmma_bf16(af.v, cur, acc[ct]);
        }
        // pipeline shape: ~36 DS reads (A 4 + 16x2), 16 WMMA
        SG_DSREAD(8);
#pragma unroll
        for (int i = 0; i < 14; ++i) { SG_WMMA(1); SG_DSREAD(2); }
        SG_WMMA(2);
    }

    // ---- phase 3: bias + relu + residual, then LN1 in registers ----
    float ps[8], ps2[8];
#pragma unroll
    for (int vr = 0; vr < 8; ++vr) { ps[vr] = 0.f; ps2[vr] = 0.f; }
#pragma unroll
    for (int ct = 0; ct < 16; ++ct) {
        int col = ct * 16 + ln;
        float bov = ldsPar[512 + col];
#pragma unroll
        for (int vr = 0; vr < 8; ++vr) {
            int rl = wave * 16 + vr + lhi * 8;
            float y = ldsY[rl * 256 + col];
            float z = y + fmaxf(acc[ct][vr] + bov, 0.f);
            acc[ct][vr] = z;
            ps[vr] += z; ps2[vr] += z * z;
        }
    }
#pragma unroll
    for (int vr = 0; vr < 8; ++vr) {
        float s  = hred_sum(ps[vr]);
        float s2 = hred_sum(ps2[vr]);
        float mu  = s * (1.f / 256.f);
        float var = s2 * (1.f / 256.f) - mu * mu;
        ps[vr]  = mu;
        ps2[vr] = rsqrtf(var + 1e-5f);
    }
#pragma unroll
    for (int ct = 0; ct < 16; ++ct) {
        int col = ct * 16 + ln;
        float g = ldsPar[768 + col], be = ldsPar[1024 + col];
#pragma unroll
        for (int vr = 0; vr < 8; ++vr) {
            float z = (acc[ct][vr] - ps[vr]) * ps2[vr] * g + be;
            out[(size_t)(row0 + wave * 16 + vr + lhi * 8) * NL + col] = z;
        }
    }
}

// =============================================================================
extern "C" void kernel_launch(void* const* d_in, const int* in_sizes, int n_in,
                              void* d_out, int out_size, void* d_ws, size_t ws_size,
                              hipStream_t stream) {
    const float* Q    = (const float*)d_in[0];
    const float* K    = (const float*)d_in[1];
    const int*   mask = (const int*)d_in[2];
    const float* Wq = (const float*)d_in[3];  const float* bq = (const float*)d_in[4];
    const float* Wk = (const float*)d_in[5];  const float* bk = (const float*)d_in[6];
    const float* Wv = (const float*)d_in[7];  const float* bv = (const float*)d_in[8];
    const float* Wo = (const float*)d_in[9];  const float* bo = (const float*)d_in[10];
    const float* g0 = (const float*)d_in[11]; const float* b0 = (const float*)d_in[12];
    const float* g1 = (const float*)d_in[13]; const float* b1 = (const float*)d_in[14];

    // workspace layout
    const size_t nElem = (size_t)ROWS * NL;           // 4,194,304
    float*  Qp    = (float*)d_ws;                     // 16 MB
    __bf16* Qhb   = (__bf16*)(Qp + nElem);            // 8 MB each
    __bf16* Khb   = Qhb + nElem;
    __bf16* Vhb   = Khb + nElem;
    float*  Xattn = (float*)(Vhb + nElem);            // 16 MB
    float*  outp  = (float*)d_out;

    size_t s1 = (size_t)(128 * 32 + 256 * 32) * sizeof(__bf16);                  // 24 KB
    proj_kernel<<<dim3(128, 3), 256, s1, stream>>>(Q, K, Wq, bq, Wk, bk, Wv, bv,
                                                   Qp, Qhb, Khb, Vhb);

    size_t s2 = (size_t)(128 * 256 + 32 * 256 + 256 * 32 + 8 * 16 * 32)
              * sizeof(__bf16);                                                  // 104 KB
    attn_kernel<<<dim3(4, NH * NB), 256, s2, stream>>>(Qhb, Khb, Vhb, Qp, mask, Xattn);

    size_t s3 = (size_t)128 * 256 * sizeof(float) + (size_t)256 * 32 * sizeof(__bf16)
              + (size_t)5 * 256 * sizeof(float);                                 // 149 KB
    mlp_ln_kernel<<<dim3(128), 256, s3, stream>>>(Xattn, Wo, bo, g0, b0, g1, b1, outp);
}
